// NegSamplerMiniBatch_52218212384858
// MI455X (gfx1250) — matrix operand order repaired and checked by reading, not compiled, not run on hardware
//
#include <hip/hip_runtime.h>

typedef __attribute__((ext_vector_type(8)))  __bf16 v8bf;
typedef __attribute__((ext_vector_type(16))) __bf16 v16bf;
typedef __attribute__((ext_vector_type(8)))  float  v8f;

#define N_EMB   16384
#define DIM     128
#define K_C     64
#define KM_ITERS 10

// ---- workspace layout (all 256B aligned) ----
static constexpr size_t OFF_BN     = 0;                                   // bf16 batch_norm   4 MB
static constexpr size_t OFF_C      = OFF_BN     + (size_t)N_EMB*DIM*2;    // f32 centroids     32 KB
static constexpr size_t OFF_CNORM  = OFF_C      + (size_t)K_C*DIM*4;      // bf16 c_norm       16 KB
static constexpr size_t OFF_LABELS = OFF_CNORM  + (size_t)K_C*DIM*2;      // i32 labels        64 KB
static constexpr size_t OFF_SUMS   = OFF_LABELS + (size_t)N_EMB*4;        // f32 sums          32 KB
static constexpr size_t OFF_COUNTS = OFF_SUMS   + (size_t)K_C*DIM*4;      // f32 counts        256 B
static constexpr size_t OFF_DONE   = OFF_COUNTS + 256;                    // i32 done flag
static constexpr size_t OFF_NEGIDX = OFF_DONE   + 256;                    // i32 neg_idx       64 KB
static constexpr size_t OFF_NC     = OFF_NEGIDX + (size_t)N_EMB*4;        // bf16 gathered     4 MB
static constexpr size_t OFF_NSI    = OFF_NC     + (size_t)N_EMB*DIM*2;    // i32 neg_sample    64 KB

static __device__ __forceinline__ unsigned short f2bf(float f) {
  unsigned u = __float_as_uint(f);
  unsigned r = u + 0x7fffu + ((u >> 16) & 1u);   // round-to-nearest-even
  return (unsigned short)(r >> 16);
}

// Load one 16x32 bf16 A tile slice (contiguous k = 16*hi + s) for k-step ks.
static __device__ __forceinline__ v16bf load_a(const unsigned short* arow, int ks, int hi) {
  return *(const v16bf*)(arow + ks * 32 + 16 * hi);
}

// ---- row L2-normalize (f32 in) -> bf16 out; one wave per 128-wide row ----
__global__ void row_norm_kernel(const float* __restrict__ in,
                                unsigned short* __restrict__ out, int nrows) {
  int wave = threadIdx.x >> 5;
  int lane = threadIdx.x & 31;
  int row  = blockIdx.x * (blockDim.x >> 5) + wave;
  if (row >= nrows) return;
  const float* p = in + (size_t)row * DIM + lane * 4;
  float x0 = p[0], x1 = p[1], x2 = p[2], x3 = p[3];
  float s = x0*x0 + x1*x1 + x2*x2 + x3*x3;
#pragma unroll
  for (int off = 16; off >= 1; off >>= 1) s += __shfl_xor(s, off, 32);
  float scale = 1.0f / fmaxf(sqrtf(s), 1e-12f);
  unsigned short* q = out + (size_t)row * DIM + lane * 4;
  q[0] = f2bf(x0 * scale); q[1] = f2bf(x1 * scale);
  q[2] = f2bf(x2 * scale); q[3] = f2bf(x3 * scale);
}

__global__ void zero_kernel(float* __restrict__ sums, float* __restrict__ counts) {
  int e = blockIdx.x * blockDim.x + threadIdx.x;
  if (e < K_C * DIM)            sums[e] = 0.0f;
  else if (e < K_C * DIM + K_C) counts[e - K_C * DIM] = 0.0f;
}

// ---- sim = bn @ cn^T (16384 x 64) fused with per-row argmax (labels) or
//      top-2 second index (neg centroid).  One wave per 16-row tile. ----
template<bool TOP2>
__global__ void sim_argmax_kernel(const unsigned short* __restrict__ bn,
                                  const unsigned short* __restrict__ cn,
                                  int* __restrict__ out_idx) {
  int wave = threadIdx.x >> 5;
  int lane = threadIdx.x & 31;
  int lo = lane & 15, hi = lane >> 4;
  int tile = blockIdx.x * (blockDim.x >> 5) + wave;
  int i0 = tile * 16;

  v16bf a[4];
  const unsigned short* arow = bn + (size_t)(i0 + lo) * DIM;
#pragma unroll
  for (int ks = 0; ks < 4; ++ks) a[ks] = load_a(arow, ks, hi);

  float b1v[8]; int b1i[8]; float b2v[8]; int b2i[8];
#pragma unroll
  for (int r = 0; r < 8; ++r) { b1v[r] = -1e38f; b1i[r] = 0x7fffffff;
                                b2v[r] = -1e38f; b2i[r] = 0x7fffffff; }

#pragma unroll
  for (int nt = 0; nt < 4; ++nt) {
    const unsigned short* brow = cn + (size_t)(nt * 16 + lo) * DIM;
    v8f acc = {};
#pragma unroll
    for (int ks = 0; ks < 4; ++ks) {
      v16bf b = *(const v16bf*)(brow + ks * 32 + 16 * hi);
      acc = __builtin_amdgcn_wmma_f32_16x16x32_bf16(false, a[ks], false, b,
                                                    (short)0, acc, false, false);
    }
    int n = nt * 16 + lo;
    // n strictly increases per lane -> strict '>' == first-occurrence argmax
#pragma unroll
    for (int r = 0; r < 8; ++r) {
      float v = acc[r];
      if (v > b1v[r]) {
        if (TOP2) { b2v[r] = b1v[r]; b2i[r] = b1i[r]; }
        b1v[r] = v; b1i[r] = n;
      } else if (TOP2 && v > b2v[r]) {
        b2v[r] = v; b2i[r] = n;
      }
    }
  }

#pragma unroll
  for (int r = 0; r < 8; ++r) {
    float v1 = b1v[r]; int i1 = b1i[r]; float v2 = b2v[r]; int i2 = b2i[r];
#pragma unroll
    for (int off = 8; off >= 1; off >>= 1) {
      float ov1 = __shfl_xor(v1, off, 16);
      int   oi1 = __shfl_xor(i1, off, 16);
      float ov2 = 0.0f; int oi2 = 0;
      if (TOP2) { ov2 = __shfl_xor(v2, off, 16); oi2 = __shfl_xor(i2, off, 16); }
      if (ov1 > v1 || (ov1 == v1 && oi1 < i1)) {
        if (TOP2) {
          if (v1 > ov2 || (v1 == ov2 && i1 < oi2)) { v2 = v1; i2 = i1; }
          else                                     { v2 = ov2; i2 = oi2; }
        }
        v1 = ov1; i1 = oi1;
      } else if (TOP2) {
        if (ov1 > v2 || (ov1 == v2 && oi1 < i2)) { v2 = ov1; i2 = oi1; }
      }
    }
    if (lo == 0) out_idx[i0 + r + 8 * hi] = TOP2 ? i2 : i1;
  }
}

// ---- segment sums via LDS (dims private per thread -> plain adds) ----
__global__ void segsum_kernel(const float* __restrict__ emb,
                              const int* __restrict__ labels,
                              float* __restrict__ sums, float* __restrict__ counts) {
  __shared__ float ls[K_C * DIM];
  __shared__ float lc[K_C];
  int t = threadIdx.x;                       // 0..127 (= dim)
  for (int k = t; k < K_C * DIM; k += 128) ls[k] = 0.0f;
  if (t < K_C) lc[t] = 0.0f;
  __syncthreads();
  int base = blockIdx.x * 128;
  for (int r = 0; r < 128; ++r) {
    int row = base + r;
    int lab = labels[row];
    float v = emb[(size_t)row * DIM + t];
    ls[lab * DIM + t] += v;                  // thread-private column -> no atomic
    if (t == 0) lc[lab] += 1.0f;
  }
  __syncthreads();
  for (int k = t; k < K_C * DIM; k += 128) atomicAdd(&sums[k], ls[k]);
  if (t < K_C) atomicAdd(&counts[t], lc[t]);
}

// ---- momentum update + convergence gate (single block) ----
__global__ void update_kernel(float* __restrict__ c, const float* __restrict__ sums,
                              const float* __restrict__ counts, int* __restrict__ done) {
  __shared__ float upd[K_C * DIM];
  __shared__ float rowsq[K_C];
  __shared__ int sel;
  int t = threadIdx.x;
  if (t < K_C) rowsq[t] = 0.0f;
  __syncthreads();
  for (int e = t; e < K_C * DIM; e += blockDim.x) {
    int row = e >> 7;
    float cf  = c[e];
    float cnt = counts[row];
    float mean = (cnt > 0.0f) ? (sums[e] / fmaxf(cnt, 1.0f)) : cf;
    float u = 0.1f * cf + 0.9f * mean;
    upd[e] = u;
    float d = cf - u;
    atomicAdd(&rowsq[row], d * d);
  }
  __syncthreads();
  if (t == 0) {
    int conv = 1;
    for (int r = 0; r < K_C; ++r) conv &= (rowsq[r] < 1e-8f) ? 1 : 0;  // TOL^2
    int s = (*done) | conv;
    *done = s;
    sel = s;
  }
  __syncthreads();
  if (!sel)
    for (int e = t; e < K_C * DIM; e += blockDim.x) c[e] = upd[e];
}

// ---- gather normalized centroid rows by neg_idx (dword copies) ----
__global__ void gather_nc_kernel(const unsigned short* __restrict__ cn,
                                 const int* __restrict__ neg_idx,
                                 unsigned short* __restrict__ nc) {
  int e = blockIdx.x * blockDim.x + threadIdx.x;   // dword index, 64 per row
  int i = e >> 6;
  int d = e & 63;
  int src = neg_idx[i];
  ((unsigned int*)nc)[e] = ((const unsigned int*)cn)[src * 64 + d];
}

// ---- column scan for the fused GEMM+argmax.  MASK specialization handles the
//      (at most two) diagonal tiles; the common path has zero diag logic. ----
template<bool MASK>
static __device__ __forceinline__ void scan_cols(
    const unsigned short* __restrict__ bn,
    const v16bf a0[4], const v16bf a1[4],
    int j0beg, int j0end, int i0, int lo, int hi,
    float* __restrict__ bv, int* __restrict__ bi)
{
  for (int j0 = j0beg; j0 < j0end; j0 += 16) {
    const unsigned short* brow = bn + (size_t)(j0 + lo) * DIM;
    v8f acc0 = {}, acc1 = {};
#pragma unroll
    for (int ks = 0; ks < 4; ++ks) {
      v16bf b = *(const v16bf*)(brow + ks * 32 + 16 * hi);
      acc0 = __builtin_amdgcn_wmma_f32_16x16x32_bf16(false, a0[ks], false, b,
                                                     (short)0, acc0, false, false);
      acc1 = __builtin_amdgcn_wmma_f32_16x16x32_bf16(false, a1[ks], false, b,
                                                     (short)0, acc1, false, false);
    }
    int j = j0 + lo;
    // j strictly increases per lane -> strict '>' == first-occurrence argmax
#pragma unroll
    for (int r = 0; r < 8; ++r) {
      float v0 = acc0[r];
      float v1 = acc1[r];
      if (MASK) {
        int m0 = r + 8 * hi;
        if (j == i0 + m0)      v0 = -1e30f;   // diagonal = -inf
        if (j == i0 + 16 + m0) v1 = -1e30f;
      }
      if (v0 > bv[r])     { bv[r]     = v0; bi[r]     = j; }
      if (v1 > bv[8 + r]) { bv[8 + r] = v1; bi[8 + r] = j; }
    }
  }
}

// ---- fused sim2 = nc @ bn^T (16384x16384) + diag-mask + per-row argmax ----
// Block = 4 waves, 32 rows (two A tiles); each wave scans 4096 columns.
// Each loaded B tile feeds two accumulators (8 WMMA per 8 b128 loads).
__global__ void __launch_bounds__(128, 1)
neg_argmax_kernel(const unsigned short* __restrict__ nc,
                  const unsigned short* __restrict__ bn,
                  int* __restrict__ nsi) {
  __shared__ float rv[128];
  __shared__ int   ri[128];
  int wave = threadIdx.x >> 5;                 // 0..3
  int lane = threadIdx.x & 31;
  int lo = lane & 15, hi = lane >> 4;
  int i0 = blockIdx.x * 32;

  v16bf a0[4], a1[4];
  const unsigned short* arow0 = nc + (size_t)(i0 + lo) * DIM;
  const unsigned short* arow1 = nc + (size_t)(i0 + 16 + lo) * DIM;
#pragma unroll
  for (int ks = 0; ks < 4; ++ks) { a0[ks] = load_a(arow0, ks, hi);
                                   a1[ks] = load_a(arow1, ks, hi); }

  float bv[16]; int bi[16];
#pragma unroll
  for (int r = 0; r < 16; ++r) { bv[r] = -1e38f; bi[r] = 0x7fffffff; }

  int jbeg = wave * (N_EMB / 4);
  int jend = jbeg + (N_EMB / 4);
  if (i0 >= jbeg && i0 < jend) {
    // this wave's strip contains the diagonal tiles [i0, i0+32): peel them
    scan_cols<false>(bn, a0, a1, jbeg,    i0,      i0, lo, hi, bv, bi);
    scan_cols<true> (bn, a0, a1, i0,      i0 + 32, i0, lo, hi, bv, bi);
    scan_cols<false>(bn, a0, a1, i0 + 32, jend,    i0, lo, hi, bv, bi);
  } else {
    scan_cols<false>(bn, a0, a1, jbeg, jend, i0, lo, hi, bv, bi);
  }

#pragma unroll
  for (int rr = 0; rr < 16; ++rr) {
    float v = bv[rr]; int idx = bi[rr];
#pragma unroll
    for (int off = 8; off >= 1; off >>= 1) {
      float ov = __shfl_xor(v, off, 16);
      int   oi = __shfl_xor(idx, off, 16);
      if (ov > v || (ov == v && oi < idx)) { v = ov; idx = oi; }
    }
    if (lo == 0) {
      int m = (rr & 7) + 8 * hi + 16 * (rr >> 3);     // 0..31 within block
      rv[wave * 32 + m] = v;
      ri[wave * 32 + m] = idx;
    }
  }
  __syncthreads();
  if (threadIdx.x < 32) {
    int m = threadIdx.x;
    float v = rv[m]; int idx = ri[m];
    for (int w = 1; w < 4; ++w) {
      float ov = rv[w * 32 + m]; int oi = ri[w * 32 + m];
      if (ov > v || (ov == v && oi < idx)) { v = ov; idx = oi; }
    }
    nsi[i0 + m] = idx;
  }
}

// ---- write outputs: centroids then gathered negatives ----
__global__ void finalize_kernel(const float* __restrict__ c, const float* __restrict__ emb,
                                const int* __restrict__ nsi, float* __restrict__ out) {
  int e = blockIdx.x * blockDim.x + threadIdx.x;
  const int total = K_C * DIM + N_EMB * DIM;
  if (e >= total) return;
  if (e < K_C * DIM) out[e] = c[e];
  else {
    int idx = e - K_C * DIM;
    int i = idx >> 7;
    int d = idx & 127;
    out[e] = emb[(size_t)nsi[i] * DIM + d];
  }
}

extern "C" void kernel_launch(void* const* d_in, const int* in_sizes, int n_in,
                              void* d_out, int out_size, void* d_ws, size_t ws_size,
                              hipStream_t stream) {
  const float* emb   = (const float*)d_in[0];
  const float* prevc = (const float*)d_in[1];
  float* out = (float*)d_out;
  char* ws = (char*)d_ws;

  unsigned short* bn   = (unsigned short*)(ws + OFF_BN);
  float*          cwk  = (float*)(ws + OFF_C);
  unsigned short* cn   = (unsigned short*)(ws + OFF_CNORM);
  int*            lab  = (int*)(ws + OFF_LABELS);
  float*          sums = (float*)(ws + OFF_SUMS);
  float*          cnts = (float*)(ws + OFF_COUNTS);
  int*            done = (int*)(ws + OFF_DONE);
  int*            nidx = (int*)(ws + OFF_NEGIDX);
  unsigned short* nc   = (unsigned short*)(ws + OFF_NC);
  int*            nsi  = (int*)(ws + OFF_NSI);

  hipMemcpyAsync(cwk, prevc, (size_t)K_C * DIM * sizeof(float),
                 hipMemcpyDeviceToDevice, stream);
  hipMemsetAsync(done, 0, sizeof(int), stream);

  row_norm_kernel<<<N_EMB / 8, 256, 0, stream>>>(emb, bn, N_EMB);

  for (int it = 0; it < KM_ITERS; ++it) {
    row_norm_kernel<<<K_C / 8, 256, 0, stream>>>(cwk, cn, K_C);
    zero_kernel<<<(K_C * DIM + K_C + 255) / 256, 256, 0, stream>>>(sums, cnts);
    sim_argmax_kernel<false><<<N_EMB / 16 / 8, 256, 0, stream>>>(bn, cn, lab);
    segsum_kernel<<<N_EMB / 128, 128, 0, stream>>>(emb, lab, sums, cnts);
    update_kernel<<<1, 256, 0, stream>>>(cwk, sums, cnts, done);
  }

  row_norm_kernel<<<K_C / 8, 256, 0, stream>>>(cwk, cn, K_C);
  sim_argmax_kernel<true><<<N_EMB / 16 / 8, 256, 0, stream>>>(bn, cn, nidx);
  gather_nc_kernel<<<(N_EMB * DIM / 2) / 256, 256, 0, stream>>>(cn, nidx, nc);
  neg_argmax_kernel<<<N_EMB / 32, 128, 0, stream>>>(nc, bn, nsi);
  finalize_kernel<<<(K_C * DIM + N_EMB * DIM + 255) / 256, 256, 0, stream>>>(cwk, emb, nsi, out);
}
